// Pyramid_Light_44547400794765
// MI455X (gfx1250) — matrix operand order repaired and checked by reading, not compiled, not run on hardware
//
#include <hip/hip_runtime.h>
#include <hip/hip_bf16.h>
#include <stdint.h>

// ---------------------------------------------------------------------------
// Sparse-masked 3D CNN backbone for MI455X (gfx1250).
// Heavy 3x3x3 convs run as implicit GEMM on v_wmma_f32_16x16x32_bf16
// (bf16 inputs, f32 accumulate). Everything else fp32 VALU.
//
// Input flattening assumption (101 tensors):
//   0: voxel_features (120000)   1: coors (120000x4 int32)   2: batch_size
//   params pytree (dicts sorted by key, tuples in order):
//   3..6   bn_final (g,b,mu,var)[128]
//   7..10  l0.blk2.bn1[16]  11..14 l0.blk2.bn2[16]
//   15 l0.blk2.w1(3,3,3,16,16) 16 l0.blk2.w2 17 l0.nin(1,16)
//   18 l0.w1(3,3,3,1,16) 19 l0.w2(3,3,3,16,16)
//   level base bp in {20 (l1,48), 45 (l2,96), 70 (l3,128)}:
//     bp+0..3 blk1.bn1  bp+4..7 blk1.bn2  bp+8 blk1.nin  bp+9 blk1.w1
//     bp+10 blk1.w2  bp+11..14 blk2.bn1  bp+15..18 blk2.bn2
//     bp+19 blk2.w1  bp+20 blk2.w2  bp+21..24 bn_pre
//   95 z.b[128]  96..99 z.bn (g,b,mu,var)[128]  100 z.w(128,8)
// ---------------------------------------------------------------------------

typedef __attribute__((ext_vector_type(16))) __bf16 v16bf;
typedef __attribute__((ext_vector_type(8)))  float  v8f;

#define EPS_BN   1e-4f
#define EPS_BN3D 1e-5f
#define NEGV    -1e30f

static __device__ __forceinline__ __bf16 tobf(float f) { return (__bf16)f; }

// ---------------- utility kernels ----------------

__global__ void k_zero(float* __restrict__ p, size_t n) {
  size_t i = (size_t)blockIdx.x * blockDim.x + threadIdx.x;
  size_t stride = (size_t)gridDim.x * blockDim.x;
  for (; i < n; i += stride) p[i] = 0.0f;
}

__global__ void k_scatter(const float* __restrict__ vf, const int* __restrict__ coors,
                          float* __restrict__ x0, float* __restrict__ mask,
                          int n, int H, int W) {
  int i = blockIdx.x * blockDim.x + threadIdx.x;
  if (i >= n) return;
  int d = coors[i * 4 + 1], h = coors[i * 4 + 2], w = coors[i * 4 + 3];
  int v = (d * H + h) * W + w;
  x0[v] = vf[i];
  mask[v] = 1.0f;
}

__global__ void k_relu(const float* __restrict__ x, float* __restrict__ y, size_t n) {
  size_t i = (size_t)blockIdx.x * blockDim.x + threadIdx.x;
  if (i < n) y[i] = fmaxf(x[i], 0.0f);
}

__global__ void k_bnrelu(const float* __restrict__ x, const float* __restrict__ g,
                         const float* __restrict__ bb, const float* __restrict__ mu,
                         const float* __restrict__ var, const float* __restrict__ mask,
                         float* __restrict__ y, int nvox, int C) {
  size_t n = (size_t)nvox * C;
  size_t i = (size_t)blockIdx.x * blockDim.x + threadIdx.x;
  if (i >= n) return;
  int c = (int)(i % C);
  size_t v = i / C;
  float s = g[c] * rsqrtf(var[c] + EPS_BN);
  float t = (x[i] - mu[c]) * s + bb[c];
  y[i] = fmaxf(t, 0.0f) * mask[v];
}

// 1x1x1 "network-in-network" branch: br[v,f] = sum_c x[v,c]*nw[c,f]
__global__ void k_nin(const float* __restrict__ x, const float* __restrict__ nw,
                      float* __restrict__ br, int nvox, int Cin, int Cout) {
  size_t n = (size_t)nvox * Cout;
  size_t i = (size_t)blockIdx.x * blockDim.x + threadIdx.x;
  if (i >= n) return;
  int f = (int)(i % Cout);
  size_t v = i / Cout;
  float s = 0.0f;
  for (int c = 0; c < Cin; ++c) s += x[v * Cin + c] * nw[c * Cout + f];
  br[i] = s;
}

// l0 first conv: Cin=1 -> 16, relu fused on input, mask on output.
__global__ void k_conv1to16(const float* __restrict__ x0, const float* __restrict__ w,
                            const float* __restrict__ mask, float* __restrict__ out,
                            int D, int H, int W) {
  int v = blockIdx.x * blockDim.x + threadIdx.x;
  int n = D * H * W;
  if (v >= n) return;
  int wq = v % W, hq = (v / W) % H, dq = v / (W * H);
  float acc[16];
#pragma unroll
  for (int k = 0; k < 16; ++k) acc[k] = 0.0f;
  for (int tap = 0; tap < 27; ++tap) {
    int dz = tap / 9, dy = (tap / 3) % 3, dx = tap % 3;
    int di = dq + dz - 1, hi = hq + dy - 1, wi = wq + dx - 1;
    if ((unsigned)di < (unsigned)D && (unsigned)hi < (unsigned)H && (unsigned)wi < (unsigned)W) {
      float f = fmaxf(x0[(di * H + hi) * W + wi], 0.0f);
      if (f != 0.0f) {
#pragma unroll
        for (int k = 0; k < 16; ++k) acc[k] += f * w[tap * 16 + k];
      }
    }
  }
  float mk = mask[v];
#pragma unroll
  for (int k = 0; k < 16; ++k) out[(size_t)v * 16 + k] = acc[k] * mk;
}

// ---------------- weight repack: fp32 (3,3,3,Cin,Cout) -> bf16 B-fragments ----
// Layout: wp[(((tap*KC+kc)*NT+nt)*32 + lane)*8 + j] holds bf16 pair for
// K = kc*32 + 16*(lane>>4) + 2j (+1), N = nt*16 + (lane&15). K>=Cin -> 0.
__global__ void k_repack(const float* __restrict__ w, uint32_t* __restrict__ wp,
                         int Cin, int Cout, int KC, int NT) {
  int idx = blockIdx.x * blockDim.x + threadIdx.x;
  int total = 27 * KC * NT * 32 * 8;
  if (idx >= total) return;
  int j = idx & 7;
  int lane = (idx >> 3) & 31;
  int rest = idx >> 8;
  int nt = rest % NT;
  int rest2 = rest / NT;
  int kc = rest2 % KC;
  int tap = rest2 / KC;
  int k = kc * 32 + 16 * (lane >> 4) + 2 * j;
  int nn = nt * 16 + (lane & 15);
  float v0 = (k < Cin)     ? w[((size_t)tap * Cin + k) * Cout + nn]     : 0.0f;
  float v1 = (k + 1 < Cin) ? w[((size_t)tap * Cin + k + 1) * Cout + nn] : 0.0f;
  union { __bf16 h[2]; uint32_t u; } pk;
  pk.h[0] = tobf(v0);
  pk.h[1] = tobf(v1);
  wp[idx] = pk.u;
}

// ---------------- WMMA implicit-GEMM 3x3x3 conv ----------------
// One wave (32 threads) computes 16 output voxels (along W) x a set of
// 16-wide Cout tiles.  SPLIT=false: all COUT/16 tiles per wave (A reuse).
// SPLIT=true: one tile per wave, tiles spread over blockIdx.x (occupancy for
// the small late levels where W/16 alone cannot fill the WGPs).
// out = conv(in)*mask [+ add].   in: (D,H,W,CIN) fp32, contiguous C.
template <int CIN, int COUT, bool SPLIT>
__global__ __launch_bounds__(32) void k_conv3(
    const float* __restrict__ in, const uint32_t* __restrict__ wp,
    const float* __restrict__ mask, const float* add, float* out,
    int D, int H, int W) {
  constexpr int KC = (CIN + 31) / 32;
  constexpr int NTALL = COUT / 16;
  constexpr int NTL = SPLIT ? 1 : NTALL;
  const int lane = threadIdx.x;
  const int m = lane & 15;
  const int half = lane >> 4;
  const int ntOff = SPLIT ? (int)(blockIdx.x % NTALL) : 0;
  const int wt = SPLIT ? (int)(blockIdx.x / NTALL) : (int)blockIdx.x;
  const int h = blockIdx.y, d = blockIdx.z;

  v8f acc[NTL] = {};

#pragma unroll 1
  for (int dz = 0; dz < 3; ++dz) {
    const int di = d + dz - 1;
    const bool dok = (unsigned)di < (unsigned)D;
#pragma unroll 1
    for (int dy = 0; dy < 3; ++dy) {
      const int hi = h + dy - 1;
      const bool rok = dok && (unsigned)hi < (unsigned)H;
      const size_t rowoff = rok ? (size_t)((di * H + hi)) * W * CIN : 0;
#pragma unroll 1
      for (int dx = 0; dx < 3; ++dx) {
        const int tap = (dz * 3 + dy) * 3 + dx;
        const int wi = wt * 16 + m + dx - 1;
        const bool ok = rok && (unsigned)wi < (unsigned)W;
        const float* src = in + rowoff + (size_t)(ok ? wi : 0) * CIN;

#pragma unroll 1
        for (int kc = 0; kc < KC; ++kc) {
          const int kb = kc * 32 + 16 * half;  // first K of run1 for this lane
          float r[16];
          if (ok && kb < CIN) {
            float4 q0 = *(const float4*)(src + kb);
            float4 q1 = *(const float4*)(src + kb + 4);
            r[0] = q0.x; r[1] = q0.y; r[2] = q0.z; r[3] = q0.w;
            r[4] = q1.x; r[5] = q1.y; r[6] = q1.z; r[7] = q1.w;
          } else {
#pragma unroll
            for (int e = 0; e < 8; ++e) r[e] = 0.0f;
          }
          if (ok && kb + 8 < CIN) {
            float4 q0 = *(const float4*)(src + kb + 8);
            float4 q1 = *(const float4*)(src + kb + 12);
            r[8]  = q0.x; r[9]  = q0.y; r[10] = q0.z; r[11] = q0.w;
            r[12] = q1.x; r[13] = q1.y; r[14] = q1.z; r[15] = q1.w;
          } else {
#pragma unroll
            for (int e = 8; e < 16; ++e) r[e] = 0.0f;
          }
          v16bf a;
#pragma unroll
          for (int e = 0; e < 16; ++e) a[e] = tobf(r[e]);

#pragma unroll
          for (int nt = 0; nt < NTL; ++nt) {
            const uint32_t* bw =
                wp + ((size_t)(((tap * KC + kc) * NTALL + (ntOff + nt)) * 32 + lane)) * 8;
            union { uint4 q[2]; v16bf v; } bu;
            bu.q[0] = *(const uint4*)bw;
            bu.q[1] = *(const uint4*)(bw + 4);
            acc[nt] = __builtin_amdgcn_wmma_f32_16x16x32_bf16(
                false, a, false, bu.v, (short)0, acc[nt], false, false);
          }
        }
      }
    }
  }

  // Epilogue: C/D layout — VGPR r holds (M=r, N=lane) lanes 0-15,
  // (M=r+8, N=lane-16) lanes 16-31.
  const size_t rowbase = ((size_t)d * H + h) * (size_t)W + (size_t)wt * 16;
#pragma unroll
  for (int r = 0; r < 8; ++r) {
    const int M = r + 8 * half;
    const float mk = mask[rowbase + M];
    const size_t o = (rowbase + M) * COUT + m;
#pragma unroll
    for (int nt = 0; nt < NTL; ++nt) {
      float v = acc[nt][r] * mk;
      if (add) v += add[o + (ntOff + nt) * 16];
      out[o + (ntOff + nt) * 16] = v;
    }
  }
}

// ---------------- 2x2x2 masked maxpool ----------------

__global__ void k_poolmask(const float* __restrict__ m, float* __restrict__ pm,
                           int Do, int Ho, int Wo) {
  int i = blockIdx.x * blockDim.x + threadIdx.x;
  int n = Do * Ho * Wo;
  if (i >= n) return;
  int wo = i % Wo, ho = (i / Wo) % Ho, dq = i / (Wo * Ho);
  int Hi = Ho * 2, Wi = Wo * 2;
  float best = 0.0f;
#pragma unroll
  for (int a = 0; a < 2; ++a)
#pragma unroll
    for (int b = 0; b < 2; ++b)
#pragma unroll
      for (int c = 0; c < 2; ++c) {
        int vi = ((2 * dq + a) * Hi + (2 * ho + b)) * Wi + (2 * wo + c);
        best = fmaxf(best, m[vi]);
      }
  pm[i] = best;
}

__global__ void k_poolx(const float* __restrict__ x, const float* __restrict__ m,
                        const float* __restrict__ pm, float* __restrict__ y,
                        int Do, int Ho, int Wo, int C) {
  size_t n = (size_t)Do * Ho * Wo * C;
  size_t i = (size_t)blockIdx.x * blockDim.x + threadIdx.x;
  if (i >= n) return;
  int c = (int)(i % C);
  size_t vo = i / C;
  int wo = (int)(vo % Wo), ho = (int)((vo / Wo) % Ho), dq = (int)(vo / ((size_t)Wo * Ho));
  int Hi = Ho * 2, Wi = Wo * 2;
  float best = NEGV;
#pragma unroll
  for (int a = 0; a < 2; ++a)
#pragma unroll
    for (int b = 0; b < 2; ++b)
#pragma unroll
      for (int cc = 0; cc < 2; ++cc) {
        int vi = ((2 * dq + a) * Hi + (2 * ho + b)) * Wi + (2 * wo + cc);
        float mk = m[vi];
        float val = (mk > 0.0f) ? x[(size_t)vi * C + c] : NEGV;
        best = fmaxf(best, val);
      }
  y[i] = (pm[vo] > 0.0f) ? best : 0.0f;
}

// ---------------- z head: depth contraction + BN(1e-5) + ReLU + transpose ----
// x: (8,8,8,128).  y[c,h,w] = relu(bn(sum_d x[d,h,w,c]*zw[c,d] + zb[c]))
__global__ void k_zhead(const float* __restrict__ x, const float* __restrict__ zw,
                        const float* __restrict__ zb, const float* __restrict__ g,
                        const float* __restrict__ bb, const float* __restrict__ mu,
                        const float* __restrict__ var, float* __restrict__ out) {
  int i = blockIdx.x * blockDim.x + threadIdx.x;
  if (i >= 8192) return;
  int hw = i & 63;
  int c = i >> 6;
  int h = hw >> 3, w = hw & 7;
  float s = 0.0f;
#pragma unroll
  for (int d = 0; d < 8; ++d)
    s += x[((size_t)(d * 8 + h) * 8 + w) * 128 + c] * zw[c * 8 + d];
  s += zb[c];
  float t = (s - mu[c]) * (g[c] * rsqrtf(var[c] + EPS_BN3D)) + bb[c];
  out[i] = fmaxf(t, 0.0f);  // out index = c*64 + h*8 + w  == i
}

// ---------------- host orchestration ----------------

static inline unsigned cdiv(size_t a, unsigned b) { return (unsigned)((a + b - 1) / b); }

static void run_repack(const float* w, uint32_t* wp, int cin, int cout, hipStream_t s) {
  int KC = (cin + 31) / 32, NT = cout / 16;
  int total = 27 * KC * NT * 256;
  k_repack<<<cdiv((size_t)total, 256), 256, 0, s>>>(w, wp, cin, cout, KC, NT);
}

static void run_conv(int cin, int cout, const float* in, const uint32_t* wp,
                     const float* mask, const float* add, float* out, int D, int H,
                     int W, hipStream_t s) {
  const bool split = (W <= 32);  // late levels: spread Cout tiles for occupancy
  dim3 b(32);
  dim3 g(split ? (W / 16) * (cout / 16) : (W / 16), H, D);
#define CONV_CASE(CI, CO)                                                            \
  else if (cin == CI && cout == CO) {                                                \
    if (split) k_conv3<CI, CO, true><<<g, b, 0, s>>>(in, wp, mask, add, out, D, H, W); \
    else       k_conv3<CI, CO, false><<<g, b, 0, s>>>(in, wp, mask, add, out, D, H, W);\
  }
  if (false) {}
  CONV_CASE(16, 16)
  CONV_CASE(16, 48)
  CONV_CASE(48, 48)
  CONV_CASE(48, 96)
  CONV_CASE(96, 96)
  CONV_CASE(96, 128)
  CONV_CASE(128, 128)
#undef CONV_CASE
}

extern "C" void kernel_launch(void* const* d_in, const int* in_sizes, int n_in,
                              void* d_out, int out_size, void* d_ws, size_t ws_size,
                              hipStream_t stream) {
  const size_t NG = 33554432;  // 128^3 * 16 floats (largest tensor) per buffer
  const size_t NM = 2097152;   // 128^3 floats (mask / 1-ch input)
  const size_t NWP = 262144;   // packed-weight dwords (max 27*4*8*256 = 221184)
  const size_t need = (3 * NG + 3 * NM + NWP) * 4;
  if (ws_size < need || n_in < 101) return;

  float* G0 = (float*)d_ws;
  float* G1 = G0 + NG;
  float* G2 = G1 + NG;
  float* M0 = G2 + NG;
  float* M1 = M0 + NM;
  float* X0 = M1 + NM;
  uint32_t* WP = (uint32_t*)(X0 + NM);

  auto F = [&](int i) { return (const float*)d_in[i]; };
  const int* coors = (const int*)d_in[1];
  const int nV = in_sizes[0];  // 120000

  auto bnrelu = [&](const float* x, int gi, const float* mask, float* y, size_t nvox, int C) {
    size_t n = nvox * (size_t)C;
    k_bnrelu<<<cdiv(n, 256), 256, 0, stream>>>(x, F(gi), F(gi + 1), F(gi + 2), F(gi + 3),
                                               mask, y, (int)nvox, C);
  };

  int D = 128, H = 128, W = 128;
  size_t nvox = (size_t)D * H * W;

  // scatter dense 1-ch grid + mask
  k_zero<<<2048, 256, 0, stream>>>(X0, nvox);
  k_zero<<<2048, 256, 0, stream>>>(M0, nvox);
  k_scatter<<<cdiv((size_t)nV, 256), 256, 0, stream>>>(F(0), coors, X0, M0, nV, H, W);

  // ---- l0 ----
  // t = conv(relu(x), w1)*mask -> G0
  k_conv1to16<<<cdiv(nvox, 256), 256, 0, stream>>>(X0, F(18), M0, G0, D, H, W);
  // relu(t) in place
  k_relu<<<cdiv(nvox * 16, 256), 256, 0, stream>>>(G0, G0, nvox * 16);
  // br = x * nin -> G1
  k_nin<<<cdiv(nvox * 16, 256), 256, 0, stream>>>(X0, F(17), G1, (int)nvox, 1, 16);
  // x = conv(relu_t, w2)*mask + br -> G1 (in-place addend)
  run_repack(F(19), WP, 16, 16, stream);
  run_conv(16, 16, G0, WP, M0, G1, G1, D, H, W, stream);
  // blk2 residual (identity)
  bnrelu(G1, 7, M0, G0, nvox, 16);
  run_repack(F(15), WP, 16, 16, stream);
  run_conv(16, 16, G0, WP, M0, nullptr, G2, D, H, W, stream);
  bnrelu(G2, 11, M0, G0, nvox, 16);
  run_repack(F(16), WP, 16, 16, stream);
  run_conv(16, 16, G0, WP, M0, G1, G1, D, H, W, stream);

  float* X = G1;  // current features
  float* A = G0;  // scratch
  float* B = G2;  // scratch
  float* Mc = M0; // current mask
  float* Mn = M1;
  int C = 16;

  const int bases[3] = {20, 45, 70};
  const int chs[3] = {48, 96, 128};
  for (int li = 0; li < 3; ++li) {
    const int bp = bases[li];
    const int ch = chs[li];
    // bn_pre then pool
    bnrelu(X, bp + 21, Mc, A, nvox, C);
    int Do = D / 2, Ho = H / 2, Wo = W / 2;
    size_t nvo = (size_t)Do * Ho * Wo;
    k_poolmask<<<cdiv(nvo, 256), 256, 0, stream>>>(Mc, Mn, Do, Ho, Wo);
    k_poolx<<<cdiv(nvo * C, 256), 256, 0, stream>>>(A, Mc, Mn, B, Do, Ho, Wo, C);
    { float* t = Mc; Mc = Mn; Mn = t; }
    D = Do; H = Ho; W = Wo; nvox = nvo;

    float* xb = B;   // pooled features
    float* brb = A;  // branch
    float* ub = X;   // bn-relu scratch (old full-res buffer, now free)

    // res_nin
    k_nin<<<cdiv(nvox * ch, 256), 256, 0, stream>>>(xb, F(bp + 8), brb, (int)nvox, C, ch);
    bnrelu(xb, bp + 0, Mc, ub, nvox, C);
    run_repack(F(bp + 9), WP, C, ch, stream);
    run_conv(C, ch, ub, WP, Mc, nullptr, xb, D, H, W, stream);  // xb dead -> reuse
    bnrelu(xb, bp + 4, Mc, ub, nvox, ch);
    run_repack(F(bp + 10), WP, ch, ch, stream);
    run_conv(ch, ch, ub, WP, Mc, brb, brb, D, H, W, stream);  // x = br + h  (in brb)
    // res_id
    bnrelu(brb, bp + 11, Mc, ub, nvox, ch);
    run_repack(F(bp + 19), WP, ch, ch, stream);
    run_conv(ch, ch, ub, WP, Mc, nullptr, xb, D, H, W, stream);
    bnrelu(xb, bp + 15, Mc, ub, nvox, ch);
    run_repack(F(bp + 20), WP, ch, ch, stream);
    run_conv(ch, ch, ub, WP, Mc, brb, brb, D, H, W, stream);  // x = x + h  (in brb)

    X = brb; A = xb; B = ub;
    C = ch;
  }

  // ---- final bn + pool + z head ----
  bnrelu(X, 3, Mc, A, nvox, C);  // bn_final, C=128, 16^3
  k_poolmask<<<cdiv((size_t)512, 256), 256, 0, stream>>>(Mc, Mn, 8, 8, 8);
  k_poolx<<<cdiv((size_t)512 * 128, 256), 256, 0, stream>>>(A, Mc, Mn, B, 8, 8, 8, 128);
  k_zhead<<<32, 256, 0, stream>>>(B, F(100), F(95), F(96), F(97), F(98), F(99),
                                  (float*)d_out);
  (void)in_sizes; (void)out_size;
}
